// SRNN_309237645759
// MI455X (gfx1250) — matrix operand order repaired and checked by reading, not compiled, not run on hardware
//
#include <hip/hip_runtime.h>
#include <hip/hip_bf16.h>

// ---------------- problem constants ----------------
#define SR_N_IN   128
#define SR_N_REC  512
#define SR_N_OUT  20
#define SR_N_T    1000
#define SR_N_B    64
#define SR_THR    1.0f
#define SR_ALPHA  0.9512294245007140f   // exp(-1/20)
#define SR_KAPPA  0.9512294245007140f
#define SR_NWG    8                     // persistent workgroups
#define SR_LDP    520                   // padded LDS row stride (halves): 65*16B rows

typedef __attribute__((ext_vector_type(16))) _Float16 v16h;
typedef __attribute__((ext_vector_type(8)))  float    v8f;
typedef __attribute__((ext_vector_type(4)))  unsigned v4u;
typedef __attribute__((ext_vector_type(4)))  int      v4i;

#define SR_AS1 __attribute__((address_space(1)))
#define SR_AS3 __attribute__((address_space(3)))

union ABfrag { v16h h; unsigned u[8]; };

__device__ __forceinline__ v8f vzero8() {
    v8f v;
#pragma unroll
    for (int i = 0; i < 8; ++i) v[i] = 0.0f;
    return v;
}

// ---------------- CDNA5 async global->LDS copy (ASYNCcnt path) ----------------
__device__ __forceinline__ void async_copy_b128(const _Float16* gsrc, _Float16* ldst) {
#if __has_builtin(__builtin_amdgcn_global_load_async_to_lds_b128)
    __builtin_amdgcn_global_load_async_to_lds_b128(
        (SR_AS1 v4i*)gsrc, (SR_AS3 v4i*)ldst, 0, 0);
#else
    *(v4u*)ldst = *(const v4u*)gsrc;   // sync fallback: global_load_b128 + ds_store_b128
#endif
}

__device__ __forceinline__ void async_copy_wait() {
#if __has_builtin(__builtin_amdgcn_s_wait_asynccnt)
    __builtin_amdgcn_s_wait_asynccnt(0);
#elif __has_builtin(__builtin_amdgcn_global_load_async_to_lds_b128)
    asm volatile("s_wait_asynccnt 0" ::: "memory");
#endif
}

// ---- A fragment (16x32 f16) from padded LDS matrix (stride SR_LDP halves) ----
// lane L: M = L%16 ; VGPR i: K = 2i + 16*(L/16)
__device__ __forceinline__ v16h load_a_lds(const _Float16* M, int m0, int kb, int lane) {
    ABfrag f;
    const int row = m0 + (lane & 15);
    const int hi  = lane >> 4;
    const unsigned* base = (const unsigned*)(M + row * SR_LDP + kb);
#pragma unroll
    for (int i = 0; i < 8; ++i) {
        const int pidx = i + 8 * hi;  // 2x ds_load_b128
        f.u[i] = base[pidx];
    }
    return f.h;
}

// ---- B fragment (32x16 f16) from padded LDS weight slice stored [n][k] ----
__device__ __forceinline__ v16h load_b_lds(const _Float16* W, int n0, int kb, int lane) {
    ABfrag f;
    const int n  = n0 + (lane & 15);
    const int hi = lane >> 4;
    const unsigned* base = (const unsigned*)(W + n * SR_LDP + kb + 16 * hi);
#pragma unroll
    for (int i = 0; i < 8; ++i) f.u[i] = base[i];                     // 2x ds_load_b128
    return f.h;
}

// ---- A fragment from f32 global source (convert to f16 pairs on the fly) ----
__device__ __forceinline__ v16h load_a_f32(const float* __restrict__ M, int ld,
                                           int m0, int kb, int lane) {
    ABfrag f;
    const int row = m0 + (lane & 15);
    const int hi  = lane >> 4;
    const float* base = M + row * ld + kb;
#pragma unroll
    for (int i = 0; i < 8; ++i) {
        const int k = 2 * i + 16 * hi;
        union { _Float16 h2[2]; unsigned u; } p;
        p.h2[0] = (_Float16)base[k];
        p.h2[1] = (_Float16)base[k + 1];
        f.u[i] = p.u;
    }
    return f.h;
}

// ---- B fragment (32x16 f16) from global weight stored [n][k] row-major ----
__device__ __forceinline__ v16h load_b_nk(const _Float16* __restrict__ W, int ld,
                                          int n0, int kb, int lane) {
    ABfrag f;
    const int n  = n0 + (lane & 15);
    const int hi = lane >> 4;
    const unsigned* base = (const unsigned*)(W + n * ld + kb + 16 * hi);
#pragma unroll
    for (int i = 0; i < 8; ++i) f.u[i] = base[i];
    return f.h;
}

__device__ __forceinline__ v8f wmma16(v16h a, v16h b, v8f c) {
    return __builtin_amdgcn_wmma_f32_16x16x32_f16(false, a, false, b,
                                                  (short)0, c, false, false);
}

// ---------------- grid-wide sense-reversal barrier ----------------
__device__ __forceinline__ void grid_sync(unsigned* __restrict__ cnt,
                                          unsigned* __restrict__ sense_g,
                                          unsigned& local_sense) {
    __syncthreads();
    const unsigned s = local_sense ^ 1u;
    if (threadIdx.x == 0) {
        __threadfence();  // make z stores visible device-wide
        unsigned arrived = __hip_atomic_fetch_add(cnt, 1u, __ATOMIC_ACQ_REL,
                                                  __HIP_MEMORY_SCOPE_AGENT);
        if (arrived == SR_NWG - 1) {
            __hip_atomic_store(cnt, 0u, __ATOMIC_RELAXED, __HIP_MEMORY_SCOPE_AGENT);
            __hip_atomic_store(sense_g, s, __ATOMIC_RELEASE, __HIP_MEMORY_SCOPE_AGENT);
        } else {
            while (__hip_atomic_load(sense_g, __ATOMIC_ACQUIRE,
                                     __HIP_MEMORY_SCOPE_AGENT) != s) {
                __builtin_amdgcn_s_sleep(1);
            }
        }
    }
    local_sense = s;
    __syncthreads();
}

// ---------------- prelude: small weight f16 conversion + state init ----------------
__global__ void srnn_prelude(const float* __restrict__ win,
                             const float* __restrict__ wout,
                             _Float16* __restrict__ win_h,
                             _Float16* __restrict__ wout_h,
                             _Float16* __restrict__ zbuf,
                             float* __restrict__ out,
                             unsigned* __restrict__ bar) {
    const int idx = blockIdx.x * blockDim.x + threadIdx.x;
    if (idx < SR_N_REC * SR_N_IN) win_h[idx] = (_Float16)win[idx];
    if (idx < 32 * SR_N_REC) {                    // w_out padded 20 -> 32 rows
        const int r = idx / SR_N_REC, c = idx % SR_N_REC;
        wout_h[idx] = (_Float16)((r < SR_N_OUT) ? wout[r * SR_N_REC + c] : 0.0f);
    }
    if (idx < SR_N_B * SR_N_REC) zbuf[idx] = (_Float16)0.0f;  // z ping buffer 0
    if (idx < SR_N_B * SR_N_OUT) {                // vo[t=0] = 0
        const int b = idx / SR_N_OUT, o = idx % SR_N_OUT;
        out[(long)b * SR_N_T * SR_N_OUT + o] = 0.0f;
    }
    if (idx < 2) bar[idx] = 0u;                   // barrier count + sense
}

// ---------------- persistent recurrent kernel ----------------
__global__ __launch_bounds__(256, 1) void srnn_persistent(
    const float* __restrict__ x,          // [64][1000][128] f32
    const float* __restrict__ wrec,       // [512][512] f32 (raw input)
    const _Float16* __restrict__ win_h,   // [512][128] (n,k)
    const _Float16* __restrict__ wout_h,  // [32][512]  (o,k)
    _Float16* __restrict__ zbuf,          // 2 x [64][512] ping-pong (global)
    float* __restrict__ out,              // [64][1000][20] raw vo (softmax later)
    unsigned* __restrict__ bar) {
    __shared__ _Float16 s_wrec[64 * SR_LDP];   // this WG's 64-column w_rec slice (66.5 KB)
    __shared__ _Float16 s_z[SR_N_B * SR_LDP];  // staged z, 64 padded rows (66.5 KB)

    const int wg   = blockIdx.x;          // 0..7
    const int wave = threadIdx.x >> 5;    // 0..7
    const int lane = threadIdx.x & 31;

    // recurrent C tiles: wave owns rows m0..m0+15, two column tiles (LDS-local n)
    const int m0   = (wave & 3) * 16;
    const int n0a_l = (wave >> 2) * 32;       // local column index into s_wrec
    const int n0b_l = n0a_l + 16;
    const int n0a  = wg * 64 + n0a_l;         // global column index (for z store)
    const int n0b  = n0a + 16;
    // output tile: one per WG, done by wave 0
    const int m0o = (wg & 3) * 16;
    const int n0o = (wg >> 2) * 16;

    // ---- one-time: convert this WG's w_rec slice (diag zeroed) into LDS ----
    for (int idx = threadIdx.x; idx < 64 * SR_N_REC; idx += 256) {
        const int nl = idx >> 9;              // 0..63
        const int k  = idx & 511;
        const int n  = wg * 64 + nl;
        const float v = wrec[n * SR_N_REC + k];
        s_wrec[nl * SR_LDP + k] = (_Float16)((n == k) ? 0.0f : v);
    }
    __syncthreads();

    _Float16* zb0 = zbuf;
    _Float16* zb1 = zbuf + SR_N_B * SR_N_REC;

    v8f v_acc0 = vzero8();   // membrane potential, persistent in VGPRs
    v8f v_acc1 = vzero8();
    v8f vo_acc = vzero8();   // leaky output accumulator (wave 0 only)
    unsigned local_sense = 0u;

    const int hi   = lane >> 4;
    const int ncol = lane & 15;

    for (int t = 0; t < SR_N_T - 1; ++t) {
        const _Float16* zr = (t & 1) ? zb1 : zb0;
        _Float16*       zw = (t & 1) ? zb0 : zb1;

        // ---- stage z_t (64x512 f16) global -> padded LDS via async copy ----
        for (int c = threadIdx.x; c < SR_N_B * (SR_N_REC / 8); c += 256) {
            const int row = c >> 6;           // 64 x 16B chunks per row
            const int cc  = (c & 63) * 8;
            async_copy_b128(zr + row * SR_N_REC + cc, s_z + row * SR_LDP + cc);
        }
        async_copy_wait();
        __syncthreads();

        // ---- vo_t = kappa*vo_{t-1} + z_t @ w_out^T (wave 0; t >= 1) ----
        if (wave == 0 && t > 0) {
            v8f o = vo_acc;
#pragma unroll
            for (int i = 0; i < 8; ++i) o[i] *= SR_KAPPA;
#pragma unroll 4
            for (int kb = 0; kb < SR_N_REC; kb += 32) {
                v16h a = load_a_lds(s_z, m0o, kb, lane);
                v16h b = load_b_nk(wout_h, SR_N_REC, n0o, kb, lane);
                o = wmma16(a, b, o);
            }
            vo_acc = o;
            const int ocol = n0o + ncol;
            if (ocol < SR_N_OUT) {
#pragma unroll
                for (int i = 0; i < 8; ++i) {
                    const int b = m0o + i + 8 * hi;
                    out[(long)b * SR_N_T * SR_N_OUT + (long)t * SR_N_OUT + ocol] = o[i];
                }
            }
        }

        // ---- acc = alpha*v - thr*z_prev (z_prev recomputed from local v) ----
        v8f acc0, acc1;
#pragma unroll
        for (int i = 0; i < 8; ++i) {
            const float z0 = (v_acc0[i] > SR_THR) ? 1.0f : 0.0f;
            const float z1 = (v_acc1[i] > SR_THR) ? 1.0f : 0.0f;
            acc0[i] = SR_ALPHA * v_acc0[i] - SR_THR * z0;
            acc1[i] = SR_ALPHA * v_acc1[i] - SR_THR * z1;
        }

        // ---- recurrent GEMM: z_t x w_rec^T, K = 512, A & B from LDS ----
#pragma unroll 4
        for (int kb = 0; kb < SR_N_REC; kb += 32) {
            v16h a  = load_a_lds(s_z, m0, kb, lane);
            v16h b0 = load_b_lds(s_wrec, n0a_l, kb, lane);
            v16h b1 = load_b_lds(s_wrec, n0b_l, kb, lane);
            acc0 = wmma16(a, b0, acc0);
            acc1 = wmma16(a, b1, acc1);
        }
        // ---- input GEMM: x_t x w_in^T, K = 128 (x hot in L2) ----
        const float* xt = x + (long)t * SR_N_IN;   // row stride N_T*N_IN
#pragma unroll
        for (int kb = 0; kb < SR_N_IN; kb += 32) {
            v16h a  = load_a_f32(xt, SR_N_T * SR_N_IN, m0, kb, lane);
            v16h b0 = load_b_nk(win_h, SR_N_IN, n0a, kb, lane);
            v16h b1 = load_b_nk(win_h, SR_N_IN, n0b, kb, lane);
            acc0 = wmma16(a, b0, acc0);
            acc1 = wmma16(a, b1, acc1);
        }
        v_acc0 = acc0;
        v_acc1 = acc1;

        // ---- publish z_{t+1} = (v_new > thr) as f16 into the write buffer ----
#pragma unroll
        for (int i = 0; i < 8; ++i) {
            const int b = m0 + i + 8 * hi;
            zw[b * SR_N_REC + n0a + ncol] = (acc0[i] > SR_THR) ? (_Float16)1.0f : (_Float16)0.0f;
            zw[b * SR_N_REC + n0b + ncol] = (acc1[i] > SR_THR) ? (_Float16)1.0f : (_Float16)0.0f;
        }

        grid_sync(bar, bar + 1, local_sense);   // all z_{t+1} visible device-wide
    }

    // ---- epilogue: vo_{999} from z_{999} (sitting in zbuf[1]) ----
    {
        const _Float16* zr = zb1;
        for (int c = threadIdx.x; c < SR_N_B * (SR_N_REC / 8); c += 256) {
            const int row = c >> 6;
            const int cc  = (c & 63) * 8;
            async_copy_b128(zr + row * SR_N_REC + cc, s_z + row * SR_LDP + cc);
        }
        async_copy_wait();
        __syncthreads();
        if (wave == 0) {
            v8f o = vo_acc;
#pragma unroll
            for (int i = 0; i < 8; ++i) o[i] *= SR_KAPPA;
#pragma unroll 4
            for (int kb = 0; kb < SR_N_REC; kb += 32) {
                v16h a = load_a_lds(s_z, m0o, kb, lane);
                v16h b = load_b_nk(wout_h, SR_N_REC, n0o, kb, lane);
                o = wmma16(a, b, o);
            }
            const int ocol = n0o + ncol;
            if (ocol < SR_N_OUT) {
#pragma unroll
                for (int i = 0; i < 8; ++i) {
                    const int b = m0o + i + 8 * hi;
                    out[(long)b * SR_N_T * SR_N_OUT + (long)(SR_N_T - 1) * SR_N_OUT + ocol] = o[i];
                }
            }
        }
    }
}

// ---------------- epilogue: softmax over the 20 outputs ----------------
__global__ void srnn_softmax(float* __restrict__ out) {
    const int r = blockIdx.x * blockDim.x + threadIdx.x;  // row = b*N_T + t
    if (r >= SR_N_B * SR_N_T) return;
    float* p = out + (long)r * SR_N_OUT;
    float m = p[0];
#pragma unroll
    for (int i = 1; i < SR_N_OUT; ++i) m = fmaxf(m, p[i]);
    float e[SR_N_OUT];
    float s = 0.0f;
#pragma unroll
    for (int i = 0; i < SR_N_OUT; ++i) { e[i] = __expf(p[i] - m); s += e[i]; }
    const float inv = 1.0f / s;
#pragma unroll
    for (int i = 0; i < SR_N_OUT; ++i) p[i] = e[i] * inv;
}

// ---------------- host launcher ----------------
extern "C" void kernel_launch(void* const* d_in, const int* in_sizes, int n_in,
                              void* d_out, int out_size, void* d_ws, size_t ws_size,
                              hipStream_t stream) {
    (void)in_sizes; (void)n_in; (void)out_size; (void)ws_size;
    const float* x     = (const float*)d_in[0];
    // d_in[1] = yt (unused by forward output)
    const float* w_in  = (const float*)d_in[2];
    const float* w_rec = (const float*)d_in[3];
    const float* w_out = (const float*)d_in[4];
    float* out = (float*)d_out;

    char* ws = (char*)d_ws;
    _Float16* win_h  = (_Float16*)(ws + 0);         // 512*128*2 = 131072
    _Float16* wout_h = (_Float16*)(ws + 131072);    //  32*512*2 =  32768
    _Float16* zbuf   = (_Float16*)(ws + 163840);    // 2*64*512*2 = 131072
    unsigned* bar    = (unsigned*)(ws + 294912);    // 2 x u32

    // prelude covers max range 512*128 = 65536 elements
    srnn_prelude<<<(SR_N_REC * SR_N_IN + 255) / 256, 256, 0, stream>>>(
        w_in, w_out, win_h, wout_h, zbuf, out, bar);

    srnn_persistent<<<SR_NWG, 256, 0, stream>>>(
        x, w_rec, win_h, wout_h, zbuf, out, bar);

    srnn_softmax<<<(SR_N_B * SR_N_T + 255) / 256, 256, 0, stream>>>(out);
}
